// SuperPointNet_gauss2_43215960932587
// MI455X (gfx1250) — matrix operand order repaired and checked by reading, not compiled, not run on hardware
//
#include <hip/hip_runtime.h>

// ---------------------------------------------------------------------------
// Types for CDNA5 WMMA
// ---------------------------------------------------------------------------
typedef __attribute__((ext_vector_type(16))) __bf16 v16bf;
typedef __attribute__((ext_vector_type(8)))  float  v8f;

union AF { v16bf v; unsigned int u[8]; };
union CF { v8f   v; float        f[8]; };

__device__ __forceinline__ unsigned short f2bf(float f) {
  union { float f; unsigned u; } c; c.f = f;
  unsigned u = c.u;
  u += 0x7fffu + ((u >> 16) & 1u);   // round-to-nearest-even
  return (unsigned short)(u >> 16);
}
__device__ __forceinline__ float bf2f(unsigned short h) {
  union { unsigned u; float f; } c; c.u = ((unsigned)h) << 16;
  return c.f;
}
__device__ __forceinline__ int imin(int a, int b) { return a < b ? a : b; }
__device__ __forceinline__ int imax(int a, int b) { return a > b ? a : b; }

// ---------------------------------------------------------------------------
// Problem constants
// ---------------------------------------------------------------------------
#define BATCH 2
#define IMH 480
#define IMW 640
#define GH  60
#define GW  80
#define MAXKP 2048
#define SORT_N 8192
#define CAND_CAP 8192

// ===========================================================================
// Parameter preparation
// ===========================================================================

// Fold conv bias + batchnorm into per-channel scale/bias:
//   y = conv*s + ((cb - m)*s + bb),  s = g / sqrt(v + eps)
__global__ void bn_fold_k(const float* __restrict__ cb, const float* __restrict__ g,
                          const float* __restrict__ bb, const float* __restrict__ m,
                          const float* __restrict__ v, float* __restrict__ scale,
                          float* __restrict__ bias, int C) {
  int c = blockIdx.x * blockDim.x + threadIdx.x;
  if (c >= 256) return;
  if (c < C) {
    float s = g[c] * rsqrtf(v[c] + 1e-5f);
    scale[c] = s;
    bias[c]  = (cb[c] - m[c]) * s + bb[c];
  } else {
    scale[c] = 0.0f;
    bias[c]  = 0.0f;
  }
}

// Pack OIHW f32 3x3 weights into WMMA B-fragment layout, bf16.
// Fragment (t, kc, nc): 256 dwords, dword index = lane*8 + d.
// B 32x16 bf16, k order used here (same as the A loads): lanes 0-15 -> K=0..15 (dword d holds K=2d,2d+1),
//                      lanes 16-31 -> K=16..31;  N = lane & 15.
__global__ void pack_w3_k(const float* __restrict__ w, unsigned int* __restrict__ wp,
                          int Cin, int Cout) {
  int KC = Cin >> 5, NC = Cout >> 4;
  int total = 9 * KC * NC * 256;
  int tid = blockIdx.x * blockDim.x + threadIdx.x;
  if (tid >= total) return;
  int d    = tid & 7;
  int lane = (tid >> 3) & 31;
  int frag = tid >> 8;
  int nc   = frag % NC;
  int rest = frag / NC;
  int kc   = rest % KC;
  int t    = rest / KC;
  int n    = nc * 16 + (lane & 15);
  int kin  = kc * 32 + ((lane >> 4) << 4) + 2 * d;
  int ky = t / 3, kx = t - 3 * (t / 3);
  size_t wi = ((size_t)n * Cin + kin) * 9 + (size_t)ky * 3 + kx;
  unsigned short lo = f2bf(w[wi]);
  unsigned short hi = f2bf(w[wi + 9]);           // next input channel
  wp[tid] = (unsigned)lo | ((unsigned)hi << 16);
}

// Pack 1x1 weights [Cout, Cin] -> fragments, N padded to Npad with zeros.
__global__ void pack_w1_k(const float* __restrict__ w, unsigned int* __restrict__ wp,
                          int Cin, int Cout, int Npad) {
  int KC = Cin >> 5, NC = Npad >> 4;
  int total = KC * NC * 256;
  int tid = blockIdx.x * blockDim.x + threadIdx.x;
  if (tid >= total) return;
  int d    = tid & 7;
  int lane = (tid >> 3) & 31;
  int frag = tid >> 8;
  int nc   = frag % NC;
  int kc   = frag / NC;
  int n    = nc * 16 + (lane & 15);
  int kin  = kc * 32 + ((lane >> 4) << 4) + 2 * d;
  unsigned int val = 0;
  if (n < Cout) {
    size_t wi = (size_t)n * Cin + kin;
    val = (unsigned)f2bf(w[wi]) | ((unsigned)f2bf(w[wi + 1]) << 16);
  }
  wp[tid] = val;
}

// ===========================================================================
// Front end
// ===========================================================================
__global__ void gray_k(const float* __restrict__ img, float* __restrict__ gray, int total) {
  int i = blockIdx.x * blockDim.x + threadIdx.x;
  if (i >= total) return;
  const int HW = IMH * IMW;
  int b = i / HW, p = i - b * HW;
  const float* base = img + (size_t)b * 3 * HW;
  gray[i] = 0.299f * base[p] + 0.587f * base[HW + p] + 0.114f * base[2 * HW + p];
}

// First conv: 1 -> 64, 3x3, fused BN + ReLU, output NHWC bf16. Tiny K, direct.
__global__ __launch_bounds__(256) void conv1_k(const float* __restrict__ gray,
                                               const float* __restrict__ w,
                                               const float* __restrict__ scale,
                                               const float* __restrict__ bias,
                                               unsigned short* __restrict__ out) {
  __shared__ float wl[576], sl[64], bl[64];
  int tid = threadIdx.x;
  for (int i = tid; i < 576; i += 256) wl[i] = w[i];
  if (tid < 64) { sl[tid] = scale[tid]; bl[tid] = bias[tid]; }
  __syncthreads();
  int gi = blockIdx.x * 256 + tid;
  const int HW = IMH * IMW;
  if (gi >= BATCH * HW) return;
  int b = gi / HW, p = gi - b * HW;
  int y = p / IMW, x = p - y * IMW;
  float g9[9];
  #pragma unroll
  for (int dy = -1; dy <= 1; ++dy)
    #pragma unroll
    for (int dx = -1; dx <= 1; ++dx) {
      int yy = y + dy, xx = x + dx;
      float v = 0.0f;
      if (yy >= 0 && yy < IMH && xx >= 0 && xx < IMW)
        v = gray[(size_t)b * HW + (size_t)yy * IMW + xx];
      g9[(dy + 1) * 3 + (dx + 1)] = v;
    }
  unsigned short* op = out + (size_t)gi * 64;
  #pragma unroll 4
  for (int c = 0; c < 64; ++c) {
    float acc = 0.0f;
    #pragma unroll
    for (int t = 0; t < 9; ++t) acc = fmaf(wl[c * 9 + t], g9[t], acc);
    op[c] = f2bf(fmaxf(acc * sl[c] + bl[c], 0.0f));
  }
}

__global__ void maxpool_k(const unsigned short* __restrict__ in, unsigned short* __restrict__ out,
                          int Ho, int Wo, int C, int total) {
  int i = blockIdx.x * blockDim.x + threadIdx.x;
  if (i >= total) return;
  int c = i % C;
  int r = i / C;
  int x = r % Wo; r /= Wo;
  int y = r % Ho; int b = r / Ho;
  int Wi = Wo * 2;
  size_t base = (((size_t)b * (Ho * 2) + 2 * y) * Wi + 2 * x) * C + c;
  float m0 = bf2f(in[base]);
  float m1 = bf2f(in[base + C]);
  float m2 = bf2f(in[base + (size_t)Wi * C]);
  float m3 = bf2f(in[base + (size_t)Wi * C + C]);
  out[i] = f2bf(fmaxf(fmaxf(m0, m1), fmaxf(m2, m3)));
}

// ===========================================================================
// 3x3 conv as implicit GEMM on v_wmma_f32_16x16x32_bf16 (+BN +ReLU)
// block = 128 threads (4 waves); block tile: 32 pixels x 64 cout.
// Each wave: one A fragment (16 px) reused across two B fragments (32 couts)
// -> 2 WMMA per K-step, half the LDS traffic per WMMA.
// Interior blocks stage the 3x34xCin patch with GLOBAL_LOAD_ASYNC_TO_LDS_B128
// (ASYNCcnt); border blocks take the scalar zero-padded path.
// ===========================================================================
__global__ __launch_bounds__(128) void conv3x3_k(const unsigned short* __restrict__ in,
                                                 const unsigned int* __restrict__ wp,
                                                 const float* __restrict__ scale,
                                                 const float* __restrict__ bias,
                                                 unsigned short* __restrict__ out,
                                                 int Hc, int Wc, int Cin, int Cout) {
  extern __shared__ unsigned int patchU[];          // 3 rows x 34 cols x Cin bf16
  int tid  = threadIdx.x;
  int lane = tid & 31;
  int wv   = tid >> 5;
  int mt   = wv >> 1;                // which 16-pixel half-strip
  int x0 = blockIdx.x << 5;          // 32-pixel strip
  int y  = blockIdx.y;
  int ngs = Cout >> 6;
  int b   = blockIdx.z / ngs;
  int ng  = blockIdx.z - b * ngs;
  int n0  = (ng << 6) + ((wv & 1) << 5);   // two 16-wide N tiles: n0, n0+16
  int Ch  = Cin >> 1;                      // dwords per pixel

  // --- stage input patch into LDS ---
  int rowBytes = 34 * Cin * 2;             // contiguous in NHWC global memory
  bool interior = (y >= 1) && (y <= Hc - 2) && (x0 >= 1) && (x0 + 34 <= Wc);
  if (interior) {
    // async memory -> LDS copy, tracked by ASYNCcnt
    unsigned ldsBase = (unsigned)(unsigned long long)(const void*)patchU;
    int rch = rowBytes >> 4;               // 16B chunks per row
    int chunks = 3 * rch;
    for (int i = tid; i < chunks; i += 128) {
      int r  = i / rch;
      int cb = (i - r * rch) << 4;
      unsigned long long g = (unsigned long long)(const void*)
          ((const char*)(in + (((size_t)b * Hc + (y - 1 + r)) * Wc + (x0 - 1)) * Cin) + cb);
      unsigned ldsAddr = ldsBase + (unsigned)(r * rowBytes + cb);
      asm volatile("global_load_async_to_lds_b128 %0, %1, off"
                   :: "v"(ldsAddr), "v"(g) : "memory");
    }
    asm volatile("s_wait_asynccnt 0x0" ::: "memory");
  } else {
    int totalu = 102 * Ch;
    for (int i = tid; i < totalu; i += 128) {
      int ci = i % Ch;
      int rc = i / Ch;
      int c = rc % 34, r = rc / 34;
      int yy = y + r - 1, xx = x0 + c - 1;
      unsigned int vL = 0;
      if (yy >= 0 && yy < Hc && xx >= 0 && xx < Wc)
        vL = ((const unsigned int*)(in + (((size_t)b * Hc + yy) * Wc + xx) * Cin))[ci];
      patchU[i] = vL;
    }
  }
  __syncthreads();

  int m  = lane & 15;
  int hi = lane >> 4;
  CF acc0, acc1;
  #pragma unroll
  for (int p = 0; p < 8; ++p) { acc0.f[p] = 0.0f; acc1.f[p] = 0.0f; }

  int KC = Cin >> 5, NC = Cout >> 4;
  int nt0 = n0 >> 4;
  for (int t = 0; t < 9; ++t) {
    int row = t / 3, colk = t - 3 * row;
    const unsigned int* lp = patchU + (row * 34 + colk + (mt << 4) + m) * Ch;
    for (int kc = 0; kc < KC; ++kc) {
      AF a;
      #pragma unroll
      for (int p = 0; p < 8; ++p) {
        int kkh = p + (hi ? 8 : 0);   // dword index within K-chunk
        a.u[p] = lp[kc * 16 + kkh];
      }
      int frag = (t * KC + kc) * NC + nt0;
      const uint4* wq4a = (const uint4*)(wp + ((size_t)frag << 8)) + (lane << 1);
      const uint4* wq4b = (const uint4*)(wp + ((size_t)(frag + 1) << 8)) + (lane << 1);
      __builtin_prefetch(wp + ((size_t)(frag + NC) << 8) + (lane << 3), 0, 1);
      uint4 q0 = wq4a[0], q1 = wq4a[1];
      uint4 q2 = wq4b[0], q3 = wq4b[1];
      AF bF0, bF1;
      bF0.u[0] = q0.x; bF0.u[1] = q0.y; bF0.u[2] = q0.z; bF0.u[3] = q0.w;
      bF0.u[4] = q1.x; bF0.u[5] = q1.y; bF0.u[6] = q1.z; bF0.u[7] = q1.w;
      bF1.u[0] = q2.x; bF1.u[1] = q2.y; bF1.u[2] = q2.z; bF1.u[3] = q2.w;
      bF1.u[4] = q3.x; bF1.u[5] = q3.y; bF1.u[6] = q3.z; bF1.u[7] = q3.w;
      acc0.v = __builtin_amdgcn_wmma_f32_16x16x32_bf16(false, a.v, false, bF0.v,
                                                       (short)0, acc0.v, false, false);
      acc1.v = __builtin_amdgcn_wmma_f32_16x16x32_bf16(false, a.v, false, bF1.v,
                                                       (short)0, acc1.v, false, false);
    }
  }
  // --- epilogue: BN + ReLU, NHWC bf16 store (guard partial strips) ---
  int nn = lane & 15;
  float sc0 = scale[n0 + nn],      bi0 = bias[n0 + nn];
  float sc1 = scale[n0 + 16 + nn], bi1 = bias[n0 + 16 + nn];
  #pragma unroll
  for (int p = 0; p < 8; ++p) {
    int mm = p + (hi << 3);
    int xp = x0 + (mt << 4) + mm;
    if (xp < Wc) {
      size_t obase = (((size_t)b * Hc + y) * Wc + xp) * Cout;
      out[obase + n0 + nn]      = f2bf(fmaxf(fmaf(acc0.f[p], sc0, bi0), 0.0f));
      out[obase + n0 + 16 + nn] = f2bf(fmaxf(fmaf(acc1.f[p], sc1, bi1), 0.0f));
    }
  }
}

// ===========================================================================
// 1x1 conv (pure GEMM) on WMMA, BN (no relu), f32 NHWC output. 1 wave / block.
// ===========================================================================
__global__ __launch_bounds__(32) void gemm1x1_k(const unsigned short* __restrict__ in,
                                                const unsigned int* __restrict__ wp,
                                                const float* __restrict__ scale,
                                                const float* __restrict__ bias,
                                                float* __restrict__ out,
                                                int K, int Npad, int Nstore) {
  int lane = threadIdx.x;
  int pix0 = blockIdx.x << 4;
  int nt   = blockIdx.y;
  int m  = lane & 15;
  int hi = lane >> 4;
  CF acc;
  #pragma unroll
  for (int p = 0; p < 8; ++p) acc.f[p] = 0.0f;
  int KC = K >> 5, Kh = K >> 1, NC = Npad >> 4;
  const unsigned int* inu = (const unsigned int*)in;
  for (int kc = 0; kc < KC; ++kc) {
    AF a;
    #pragma unroll
    for (int p = 0; p < 8; ++p) {
      int kkh = p + (hi ? 8 : 0);
      a.u[p] = inu[(size_t)(pix0 + m) * Kh + kc * 16 + kkh];
    }
    int frag = kc * NC + nt;
    const uint4* wq4 = (const uint4*)(wp + ((size_t)frag << 8)) + (lane << 1);
    uint4 q0 = wq4[0], q1 = wq4[1];
    AF bF;
    bF.u[0] = q0.x; bF.u[1] = q0.y; bF.u[2] = q0.z; bF.u[3] = q0.w;
    bF.u[4] = q1.x; bF.u[5] = q1.y; bF.u[6] = q1.z; bF.u[7] = q1.w;
    acc.v = __builtin_amdgcn_wmma_f32_16x16x32_bf16(false, a.v, false, bF.v,
                                                    (short)0, acc.v, false, false);
  }
  int n = (nt << 4) + (lane & 15);
  if (n < Nstore) {
    float sc = scale[n], bi = bias[n];
    #pragma unroll
    for (int p = 0; p < 8; ++p) {
      int mm = p + (hi << 3);
      out[(size_t)(pix0 + mm) * Nstore + n] = fmaf(acc.f[p], sc, bi);
    }
  }
}

// ===========================================================================
// Detector head post: softmax(65) drop dustbin + pixel shuffle -> [B,480,640]
// ===========================================================================
__global__ void softmax_shuffle_k(const float* __restrict__ semi, float* __restrict__ dense) {
  int i = blockIdx.x * blockDim.x + threadIdx.x;
  const int HW = GH * GW;
  if (i >= BATCH * HW) return;
  const float* s = semi + (size_t)i * 65;
  int b = i / HW, p = i - b * HW;
  int yc = p / GW, xc = p - yc * GW;
  float mx = s[0];
  for (int c = 1; c < 65; ++c) mx = fmaxf(mx, s[c]);
  float sum = 0.0f;
  for (int c = 0; c < 65; ++c) sum += __expf(s[c] - mx);
  float inv = 1.0f / sum;
  for (int c = 0; c < 64; ++c) {
    int r = c >> 3, cc = c & 7;
    dense[((size_t)b * IMH + (yc * 8 + r)) * IMW + (xc * 8 + cc)] = __expf(s[c] - mx) * inv;
  }
}

// ===========================================================================
// Descriptor normalize: write NCHW to d_out + keep NHWC normalized copy in ws
// ===========================================================================
__global__ __launch_bounds__(256) void desc_norm_k(const float* __restrict__ draw,
                                                   float* __restrict__ outNCHW,
                                                   float* __restrict__ descN) {
  __shared__ float red[256];
  int pix = blockIdx.x;
  int ch  = threadIdx.x;
  float v = draw[(size_t)pix * 256 + ch];
  red[ch] = v * v;
  __syncthreads();
  for (int s = 128; s > 0; s >>= 1) {
    if (ch < s) red[ch] += red[ch + s];
    __syncthreads();
  }
  float inv = 1.0f / sqrtf(red[0]);
  float nv = v * inv;
  const int HW = GH * GW;
  int b = pix / HW, p = pix - b * HW;
  int yc = p / GW, xc = p - yc * GW;
  outNCHW[(((size_t)b * 256 + ch) * GH + yc) * GW + xc] = nv;
  descN[(size_t)pix * 256 + ch] = nv;
}

// ===========================================================================
// simple_nms (radius 3) as iterated 7x7 maxpool / mask updates
// ===========================================================================
__device__ __forceinline__ float mp7(const float* __restrict__ m, int b, int y, int x) {
  int y0 = imax(y - 3, 0), y1 = imin(y + 3, IMH - 1);
  int x0 = imax(x - 3, 0), x1 = imin(x + 3, IMW - 1);
  float best = -1e30f;
  const float* row = m + (size_t)b * IMH * IMW;
  for (int yy = y0; yy <= y1; ++yy)
    for (int xx = x0; xx <= x1; ++xx)
      best = fmaxf(best, row[(size_t)yy * IMW + xx]);
  return best;
}

__global__ void nms_mask_k(const float* __restrict__ s, float* __restrict__ mask, int total) {
  int i = blockIdx.x * blockDim.x + threadIdx.x;
  if (i >= total) return;
  const int HW = IMH * IMW;
  int b = i / HW, p = i - b * HW;
  int y = p / IMW, x = p - y * IMW;
  mask[i] = (s[i] == mp7(s, b, y, x)) ? 1.0f : 0.0f;
}

__global__ void nms_supp_k(const float* __restrict__ s, const float* __restrict__ mask,
                           float* __restrict__ ss, float* __restrict__ supp, int total) {
  int i = blockIdx.x * blockDim.x + threadIdx.x;
  if (i >= total) return;
  const int HW = IMH * IMW;
  int b = i / HW, p = i - b * HW;
  int y = p / IMW, x = p - y * IMW;
  bool sp = mp7(mask, b, y, x) > 0.0f;
  supp[i] = sp ? 1.0f : 0.0f;
  ss[i]   = sp ? 0.0f : s[i];
}

__global__ void nms_update_k(const float* __restrict__ ss, const float* __restrict__ supp,
                             float* __restrict__ mask, int total) {
  int i = blockIdx.x * blockDim.x + threadIdx.x;
  if (i >= total) return;
  const int HW = IMH * IMW;
  int b = i / HW, p = i - b * HW;
  int y = p / IMW, x = p - y * IMW;
  bool newmax = (ss[i] == mp7(ss, b, y, x));
  if (newmax && supp[i] == 0.0f) mask[i] = 1.0f;
}

__global__ void nms_final_k(const float* __restrict__ s, const float* __restrict__ mask,
                            float* __restrict__ nms, int total) {
  int i = blockIdx.x * blockDim.x + threadIdx.x;
  if (i >= total) return;
  const int HW = IMH * IMW;
  int p = i % HW;
  int y = p / IMW, x = p - y * IMW;
  float v = (mask[i] != 0.0f) ? s[i] : 0.0f;
  if (y < 4 || y >= IMH - 4 || x < 4 || x >= IMW - 4) v = -1.0f;
  nms[i] = v;
}

// ===========================================================================
// Top-K: histogram threshold select + LDS bitonic sort (desc score, asc idx)
// ===========================================================================
__global__ void zero_u32_k(unsigned int* p, int n) {
  int i = blockIdx.x * blockDim.x + threadIdx.x;
  if (i < n) p[i] = 0;
}

__global__ void hist_k(const float* __restrict__ nms, unsigned int* __restrict__ hist, int total) {
  int i = blockIdx.x * blockDim.x + threadIdx.x;
  if (i >= total) return;
  float s = nms[i];
  if (s <= 0.0f) return;
  const int HW = IMH * IMW;
  int b = i / HW;
  int bin = imin(1023, (int)(s * 1024.0f));
  atomicAdd(&hist[b * 1024 + bin], 1u);
}

__global__ void scan_k(const unsigned int* __restrict__ hist, int* __restrict__ thrBin) {
  if (threadIdx.x != 0) return;
  int b = blockIdx.x;
  unsigned cum = 0;
  int t = 0;
  for (int i = 1023; i >= 0; --i) {
    cum += hist[b * 1024 + i];
    t = i;
    if (cum >= MAXKP) break;
  }
  thrBin[b] = t;
}

__global__ void compact_k(const float* __restrict__ nms, const int* __restrict__ thrBin,
                          float* __restrict__ cS, int* __restrict__ cI,
                          unsigned int* __restrict__ cnt, int total) {
  int i = blockIdx.x * blockDim.x + threadIdx.x;
  if (i >= total) return;
  float s = nms[i];
  if (s <= 0.0f) return;
  const int HW = IMH * IMW;
  int b = i / HW;
  int bin = imin(1023, (int)(s * 1024.0f));
  if (bin < thrBin[b]) return;
  unsigned pos = atomicAdd(&cnt[b], 1u);
  if (pos < CAND_CAP) {
    cS[(size_t)b * CAND_CAP + pos] = s;
    cI[(size_t)b * CAND_CAP + pos] = i - b * HW;
  }
}

__global__ __launch_bounds__(1024) void topk_sort_k(const float* __restrict__ cS,
                                                    const int* __restrict__ cI,
                                                    const unsigned int* __restrict__ cnt,
                                                    float* __restrict__ kp,
                                                    float* __restrict__ ks) {
  __shared__ float ss[SORT_N];
  __shared__ int   si[SORT_N];
  int b = blockIdx.x, tid = threadIdx.x;
  int n = (int)cnt[b]; if (n > CAND_CAP) n = CAND_CAP;
  for (int i = tid; i < SORT_N; i += 1024) {
    if (i < n) { ss[i] = cS[(size_t)b * CAND_CAP + i]; si[i] = cI[(size_t)b * CAND_CAP + i]; }
    else       { ss[i] = -1e30f; si[i] = 0x7fffffff; }
  }
  __syncthreads();
  for (int k = 2; k <= SORT_N; k <<= 1) {
    for (int j = k >> 1; j > 0; j >>= 1) {
      for (int i = tid; i < SORT_N; i += 1024) {
        int ixj = i ^ j;
        if (ixj > i) {
          bool desc = ((i & k) == 0);
          float s1 = ss[i], s2 = ss[ixj];
          int   i1 = si[i], i2 = si[ixj];
          bool firstBetter = (s1 > s2) || (s1 == s2 && i1 < i2);
          bool dosw = desc ? (!firstBetter) : firstBetter;
          if (dosw) { ss[i] = s2; ss[ixj] = s1; si[i] = i2; si[ixj] = i1; }
        }
      }
      __syncthreads();
    }
  }
  for (int kidx = tid; kidx < MAXKP; kidx += 1024) {
    float s = ss[kidx];
    int id  = si[kidx];
    float kx = 0.0f, ky = 0.0f;
    if (s <= -1e29f) { s = 0.0f; }
    else { kx = (float)(id % IMW); ky = (float)(id / IMW); }
    kp[((size_t)b * MAXKP + kidx) * 2 + 0] = kx;
    kp[((size_t)b * MAXKP + kidx) * 2 + 1] = ky;
    ks[(size_t)b * MAXKP + kidx] = (s > 0.005f) ? s : 0.0f;
  }
}

// ===========================================================================
// Bilinear descriptor sampling + renormalize -> [B,256,K]
// ===========================================================================
__global__ __launch_bounds__(256) void sample_k(const float* __restrict__ descN,
                                                const float* __restrict__ kp,
                                                float* __restrict__ out5) {
  __shared__ float red[256];
  int gk = blockIdx.x;
  int b = gk / MAXKP, k = gk - b * MAXKP;
  int ch = threadIdx.x;
  float x = kp[((size_t)b * MAXKP + k) * 2 + 0];
  float y = kp[((size_t)b * MAXKP + k) * 2 + 1];
  float ix = (x - 3.5f) * ((float)(GW - 1) / ((float)(GW * 8) - 4.5f));
  float iy = (y - 3.5f) * ((float)(GH - 1) / ((float)(GH * 8) - 4.5f));
  int x0 = (int)floorf(ix), y0 = (int)floorf(iy);
  float wx = ix - (float)x0, wy = iy - (float)y0;
  const float* base = descN + (size_t)b * GH * GW * 256;
  auto get = [&](int xi, int yi) -> float {
    if (xi < 0 || xi >= GW || yi < 0 || yi >= GH) return 0.0f;
    return base[((size_t)yi * GW + xi) * 256 + ch];
  };
  float v = get(x0, y0) * (1.0f - wx) * (1.0f - wy)
          + get(x0 + 1, y0) * wx * (1.0f - wy)
          + get(x0, y0 + 1) * (1.0f - wx) * wy
          + get(x0 + 1, y0 + 1) * wx * wy;
  red[ch] = v * v;
  __syncthreads();
  for (int s = 128; s > 0; s >>= 1) {
    if (ch < s) red[ch] += red[ch + s];
    __syncthreads();
  }
  float inv = 1.0f / fmaxf(sqrtf(red[0]), 1e-12f);
  out5[(((size_t)b * 256) + ch) * MAXKP + k] = v * inv;
}

// ===========================================================================
// Host orchestration
// ===========================================================================
extern "C" void kernel_launch(void* const* d_in, const int* in_sizes, int n_in,
                              void* d_out, int out_size, void* d_ws, size_t ws_size,
                              hipStream_t stream) {
  (void)in_sizes; (void)n_in; (void)out_size; (void)ws_size;
#define IN(i) ((const float*)d_in[i])

  // ---- workspace carve ----
  char* ws = (char*)d_ws;
  size_t off = 0;
  auto A = [&](size_t bytes) -> char* {
    char* p = ws + off;
    off += (bytes + 255) & ~(size_t)255;
    return p;
  };
  float*          gray    = (float*)A((size_t)BATCH * IMH * IMW * 4);
  unsigned short* actA    = (unsigned short*)A((size_t)BATCH * IMH * IMW * 64 * 2);
  unsigned short* actB    = (unsigned short*)A((size_t)BATCH * IMH * IMW * 64 * 2);
  unsigned int*   wp_inc2 = (unsigned int*)A((size_t)9 * 64 * 64 * 2);
  unsigned int*   wp_d1a  = (unsigned int*)A((size_t)9 * 64 * 64 * 2);
  unsigned int*   wp_d1b  = (unsigned int*)A((size_t)9 * 64 * 64 * 2);
  unsigned int*   wp_d2a  = (unsigned int*)A((size_t)9 * 64 * 128 * 2);
  unsigned int*   wp_d2b  = (unsigned int*)A((size_t)9 * 128 * 128 * 2);
  unsigned int*   wp_d3a  = (unsigned int*)A((size_t)9 * 128 * 128 * 2);
  unsigned int*   wp_d3b  = (unsigned int*)A((size_t)9 * 128 * 128 * 2);
  unsigned int*   wp_Pa   = (unsigned int*)A((size_t)9 * 128 * 256 * 2);
  unsigned int*   wp_Da   = (unsigned int*)A((size_t)9 * 128 * 256 * 2);
  unsigned int*   wp_Pb   = (unsigned int*)A((size_t)256 * 80 * 2);
  unsigned int*   wp_Db   = (unsigned int*)A((size_t)256 * 256 * 2);
  float*          scb     = (float*)A((size_t)12 * 256 * 4);
  float*          bib     = (float*)A((size_t)12 * 256 * 4);
  float*          semi    = (float*)A((size_t)BATCH * GH * GW * 65 * 4);
  float*          descRaw = (float*)A((size_t)BATCH * GH * GW * 256 * 4);
  float*          descN   = (float*)A((size_t)BATCH * GH * GW * 256 * 4);
  float*          maskB   = (float*)A((size_t)BATCH * IMH * IMW * 4);
  float*          ssB     = (float*)A((size_t)BATCH * IMH * IMW * 4);
  float*          suppB   = (float*)A((size_t)BATCH * IMH * IMW * 4);
  unsigned int*   hist    = (unsigned int*)A((size_t)BATCH * 1024 * 4);
  int*            thrBin  = (int*)A((size_t)BATCH * 4);
  float*          candS   = (float*)A((size_t)BATCH * CAND_CAP * 4);
  int*            candI   = (int*)A((size_t)BATCH * CAND_CAP * 4);
  unsigned int*   candCnt = (unsigned int*)A((size_t)BATCH * 4);

  // ---- output regions ----
  float* r0 = (float*)d_out;                      // dense_scores [2,480,640]
  float* r1 = r0 + (size_t)BATCH * IMH * IMW;     // nms          [2,480,640]
  float* r2 = r1 + (size_t)BATCH * IMH * IMW;     // dense_desc   [2,256,60,80]
  float* r3 = r2 + (size_t)BATCH * 256 * GH * GW; // keypoints    [2,2048,2]
  float* r4 = r3 + (size_t)BATCH * MAXKP * 2;     // kp_scores    [2,2048]
  float* r5 = r4 + (size_t)BATCH * MAXKP;         // sampled_desc [2,256,2048]

  auto cdiv = [](int a, int b) { return (a + b - 1) / b; };

  // ---- fold conv bias + BN into scale/bias (slots) ----
  struct { int cb, g, bb, m, v, C, slot; } BN[12] = {
    { 2,  3,  4,  5,  6,  64, 0},  // inc.bn1
    { 8,  9, 10, 11, 12,  64, 1},  // inc.bn2
    {14, 15, 16, 17, 18,  64, 2},  // down1.bn1
    {20, 21, 22, 23, 24,  64, 3},  // down1.bn2
    {26, 27, 28, 29, 30, 128, 4},  // down2.bn1
    {32, 33, 34, 35, 36, 128, 5},  // down2.bn2
    {38, 39, 40, 41, 42, 128, 6},  // down3.bn1
    {44, 45, 46, 47, 48, 128, 7},  // down3.bn2
    {50, 51, 52, 53, 54, 256, 8},  // bnPa
    {56, 57, 58, 59, 60,  65, 9},  // bnPb
    {62, 63, 64, 65, 66, 256, 10}, // bnDa
    {68, 69, 70, 71, 72, 256, 11}, // bnDb
  };
  for (int i = 0; i < 12; ++i)
    bn_fold_k<<<1, 256, 0, stream>>>(IN(BN[i].cb), IN(BN[i].g), IN(BN[i].bb),
                                     IN(BN[i].m), IN(BN[i].v),
                                     scb + BN[i].slot * 256, bib + BN[i].slot * 256, BN[i].C);
  auto SC = [&](int s) { return scb + s * 256; };
  auto BI = [&](int s) { return bib + s * 256; };

  // ---- pack weights into WMMA fragment layout ----
  struct { int wi, cin, cout; unsigned int* dst; } W3[9] = {
    { 7,  64,  64, wp_inc2}, {13, 64, 64, wp_d1a}, {19, 64, 64, wp_d1b},
    {25,  64, 128, wp_d2a},  {31, 128, 128, wp_d2b},
    {37, 128, 128, wp_d3a},  {43, 128, 128, wp_d3b},
    {49, 128, 256, wp_Pa},   {61, 128, 256, wp_Da},
  };
  for (int i = 0; i < 9; ++i) {
    int total = 9 * (W3[i].cin >> 5) * (W3[i].cout >> 4) * 256;
    pack_w3_k<<<cdiv(total, 256), 256, 0, stream>>>(IN(W3[i].wi), W3[i].dst, W3[i].cin, W3[i].cout);
  }
  pack_w1_k<<<cdiv((256 >> 5) * (80 >> 4) * 256, 256), 256, 0, stream>>>(IN(55), wp_Pb, 256, 65, 80);
  pack_w1_k<<<cdiv((256 >> 5) * (256 >> 4) * 256, 256), 256, 0, stream>>>(IN(67), wp_Db, 256, 256, 256);

  const int HWtot = BATCH * IMH * IMW;

  // ---- backbone ----
  gray_k<<<cdiv(HWtot, 256), 256, 0, stream>>>(IN(0), gray, HWtot);
  conv1_k<<<cdiv(HWtot, 256), 256, 0, stream>>>(gray, IN(1), SC(0), BI(0), actA);

  auto conv3 = [&](const unsigned short* in, unsigned int* wpk, int slot,
                   unsigned short* out, int Hc, int Wc, int Cin, int Cout) {
    dim3 g(cdiv(Wc, 32), Hc, BATCH * (Cout / 64));
    conv3x3_k<<<g, 128, (size_t)3 * 34 * Cin * 2, stream>>>(in, wpk, SC(slot), BI(slot), out,
                                                            Hc, Wc, Cin, Cout);
  };
  auto pool = [&](const unsigned short* in, unsigned short* out, int Ho, int Wo, int C) {
    int total = BATCH * Ho * Wo * C;
    maxpool_k<<<cdiv(total, 256), 256, 0, stream>>>(in, out, Ho, Wo, C, total);
  };

  conv3(actA, wp_inc2, 1, actB, 480, 640, 64, 64);   // x1 = actB
  pool(actB, actA, 240, 320, 64);
  conv3(actA, wp_d1a, 2, actB, 240, 320, 64, 64);
  conv3(actB, wp_d1b, 3, actA, 240, 320, 64, 64);    // x2 = actA
  pool(actA, actB, 120, 160, 64);
  conv3(actB, wp_d2a, 4, actA, 120, 160, 64, 128);
  conv3(actA, wp_d2b, 5, actB, 120, 160, 128, 128);  // x3 = actB
  pool(actB, actA, 60, 80, 128);
  conv3(actA, wp_d3a, 6, actB, 60, 80, 128, 128);
  conv3(actB, wp_d3b, 7, actA, 60, 80, 128, 128);    // x4 = actA

  // ---- detector head ----
  conv3(actA, wp_Pa, 8, actB, 60, 80, 128, 256);     // cPa
  {
    dim3 g((BATCH * GH * GW) / 16, 80 / 16);
    gemm1x1_k<<<g, 32, 0, stream>>>(actB, wp_Pb, SC(9), BI(9), semi, 256, 80, 65);
  }
  softmax_shuffle_k<<<cdiv(BATCH * GH * GW, 256), 256, 0, stream>>>(semi, r0);

  // ---- descriptor head ----
  conv3(actA, wp_Da, 10, actB, 60, 80, 128, 256);    // cDa
  {
    dim3 g((BATCH * GH * GW) / 16, 256 / 16);
    gemm1x1_k<<<g, 32, 0, stream>>>(actB, wp_Db, SC(11), BI(11), descRaw, 256, 256, 256);
  }
  desc_norm_k<<<BATCH * GH * GW, 256, 0, stream>>>(descRaw, r2, descN);

  // ---- NMS (2 suppression rounds) + border ----
  int nb = cdiv(HWtot, 256);
  nms_mask_k<<<nb, 256, 0, stream>>>(r0, maskB, HWtot);
  for (int it = 0; it < 2; ++it) {
    nms_supp_k<<<nb, 256, 0, stream>>>(r0, maskB, ssB, suppB, HWtot);
    nms_update_k<<<nb, 256, 0, stream>>>(ssB, suppB, maskB, HWtot);
  }
  nms_final_k<<<nb, 256, 0, stream>>>(r0, maskB, r1, HWtot);

  // ---- top-K keypoints ----
  zero_u32_k<<<cdiv(BATCH * 1024, 256), 256, 0, stream>>>(hist, BATCH * 1024);
  zero_u32_k<<<1, 32, 0, stream>>>(candCnt, BATCH);
  hist_k<<<nb, 256, 0, stream>>>(r1, hist, HWtot);
  scan_k<<<BATCH, 1, 0, stream>>>(hist, thrBin);
  compact_k<<<nb, 256, 0, stream>>>(r1, thrBin, candS, candI, candCnt, HWtot);
  topk_sort_k<<<BATCH, 1024, 0, stream>>>(candS, candI, candCnt, r3, r4);

  // ---- descriptor sampling at keypoints ----
  sample_k<<<BATCH * MAXKP, 256, 0, stream>>>(descN, r3, r5);
#undef IN
}
